// TernaryMLP_20693152432262
// MI455X (gfx1250) — matrix operand order repaired and checked by reading, not compiled, not run on hardware
//
#include <hip/hip_runtime.h>

typedef __attribute__((ext_vector_type(16))) _Float16 v16h;
typedef __attribute__((ext_vector_type(8)))  _Float16 v8h;
typedef __attribute__((ext_vector_type(4)))  _Float16 v4h;
typedef __attribute__((ext_vector_type(8)))  float    v8f;
typedef __attribute__((ext_vector_type(4)))  float    v4f;
typedef __attribute__((ext_vector_type(4)))  int      v4i;
typedef __attribute__((ext_vector_type(8)))  int      v8i;
typedef __attribute__((ext_vector_type(4)))  unsigned v4u;

#define TERN_THRESH 0.75f

static constexpr int DIM_IN  = 4096;
static constexpr int DIM_H   = 8192;
static constexpr int DIM_OUT = 4096;
static constexpr int DIM_N   = 8192;

#if __has_builtin(__builtin_amdgcn_tensor_load_to_lds)
#define USE_TDM 1
#else
#define USE_TDM 0
#endif

// ---------------------------------------------------------------------------
// CDNA5 async global->LDS copy (ASYNCcnt) — fallback staging path
// ---------------------------------------------------------------------------
__device__ __forceinline__ void async_load_b128(const void* g, void* l) {
#if __has_builtin(__builtin_amdgcn_global_load_async_to_lds_b128)
  __builtin_amdgcn_global_load_async_to_lds_b128(
      (__attribute__((address_space(1))) v4i*)g,
      (__attribute__((address_space(3))) v4i*)l, 0, 0);
#else
  unsigned lofs = (unsigned)(unsigned long)(__attribute__((address_space(3))) void*)l;
  asm volatile("global_load_async_to_lds_b128 %0, %1, off"
               :: "v"(lofs), "v"(g) : "memory");
#endif
}

__device__ __forceinline__ void wait_async0() {
#if __has_builtin(__builtin_amdgcn_s_wait_asynccnt)
  __builtin_amdgcn_s_wait_asynccnt(0);
#else
  asm volatile("s_wait_asynccnt 0x0" ::: "memory");
#endif
}

__device__ __forceinline__ void wait_tensor0() {
#if __has_builtin(__builtin_amdgcn_s_wait_tensorcnt)
  __builtin_amdgcn_s_wait_tensorcnt(0);
#else
  asm volatile("s_wait_tensorcnt 0x0" ::: "memory");
#endif
}

#if USE_TDM
// ---------------------------------------------------------------------------
// TDM: one instruction DMAs a rows x 32-half 2D tile (64B rows) from global
// into LDS with +16B padding after every 64B -> padded row stride 80B (LD=40).
// D# packing per CDNA5 ISA ch.8:
//   group0: [1:0]=count=1, [63:32]=lds_addr, [120:64]=global_addr, [127:126]=2
//   group1: data_size=1(2B), pad_enable, pad_interval=3(64B), pad_amount=3(16B),
//           tensor_dim0=K, tensor_dim1=rows, tile_dim0=32, tile_dim1=rows,
//           tensor_dim0_stride=K
// ---------------------------------------------------------------------------
__device__ __forceinline__ void tdm_load_tile(const _Float16* gbase, void* lds,
                                              int rows, int K) {
  unsigned long ga = (unsigned long)gbase;
  unsigned laddr =
      (unsigned)(unsigned long)(__attribute__((address_space(3))) void*)lds;
  v4u g0;
  g0[0] = 1u;                                            // count=1
  g0[1] = laddr;                                         // lds_addr
  g0[2] = (unsigned)(ga & 0xFFFFFFFFu);                  // global_addr lo
  g0[3] = (unsigned)((ga >> 32) & 0x01FFFFFFu) | (2u << 30);  // ga hi | type=2
  unsigned uk = (unsigned)K, ur = (unsigned)rows;
  v8i g1;
  g1[0] = (int)((1u << 16) | (1u << 20) | (3u << 22) | (3u << 25));
  g1[1] = (int)((uk & 0xFFFFu) << 16);                   // tensor_dim0 lo16
  g1[2] = (int)(((uk >> 16) & 0xFFFFu) | ((ur & 0xFFFFu) << 16));
  g1[3] = (int)(32u << 16);                              // tile_dim0 = 32
  g1[4] = (int)(ur & 0xFFFFu);                           // tile_dim1 = rows
  g1[5] = (int)uk;                                       // dim0_stride lo32
  g1[6] = 0;
  g1[7] = 0;
  v4i z4 = {0, 0, 0, 0};                                 // groups 2/3 unused (2D)
  v8i z8 = {0, 0, 0, 0, 0, 0, 0, 0};
  __builtin_amdgcn_tensor_load_to_lds(g0, g1, z4, z4, z8, 0);
}
#endif

// ---------------------------------------------------------------------------
// Pass 1: sum of |w| over the whole matrix -> ws scalar (f32 atomic)
// ---------------------------------------------------------------------------
__global__ void absum_kernel(const float* __restrict__ w, long n,
                             float* __restrict__ out_sumabs) {
  __shared__ float red[256];
  float s = 0.f;
  long stride = (long)gridDim.x * 256L;
  for (long i = blockIdx.x * 256L + threadIdx.x; i < n; i += stride)
    s += fabsf(w[i]);
  red[threadIdx.x] = s;
  __syncthreads();
  for (int off = 128; off > 0; off >>= 1) {
    if ((int)threadIdx.x < off) red[threadIdx.x] += red[threadIdx.x + off];
    __syncthreads();
  }
  if (threadIdx.x == 0) atomicAdd(out_sumabs, red[0]);
}

// ---------------------------------------------------------------------------
// Pass 2: ternarize to f16 {-1,0,1} + accumulate masked |w| sum and count.
// ---------------------------------------------------------------------------
__global__ void quantize_kernel(const float* __restrict__ w, long n,
                                const float* __restrict__ sumabs,
                                _Float16* __restrict__ wq,
                                float* __restrict__ out_msum,
                                unsigned* __restrict__ out_mcnt) {
  const float delta = TERN_THRESH * (*sumabs) / (float)n;
  __shared__ float    reds[256];
  __shared__ unsigned redc[256];
  float s = 0.f;
  unsigned c = 0u;
  long stride = (long)gridDim.x * 256L;
  for (long i = blockIdx.x * 256L + threadIdx.x; i < n; i += stride) {
    float x  = w[i];
    float ax = fabsf(x);
    bool  m  = ax > delta;
    s += m ? ax : 0.f;
    c += m ? 1u : 0u;
    float q = m ? (x > 0.f ? 1.f : -1.f) : 0.f;
    wq[i] = (_Float16)q;
  }
  reds[threadIdx.x] = s;
  redc[threadIdx.x] = c;
  __syncthreads();
  for (int off = 128; off > 0; off >>= 1) {
    if ((int)threadIdx.x < off) {
      reds[threadIdx.x] += reds[threadIdx.x + off];
      redc[threadIdx.x] += redc[threadIdx.x + off];
    }
    __syncthreads();
  }
  if (threadIdx.x == 0) {
    atomicAdd(out_msum, reds[0]);
    atomicAdd(out_mcnt, redc[0]);
  }
}

// ---------------------------------------------------------------------------
// f32 -> f16 elementwise convert (vectorized x4)
// ---------------------------------------------------------------------------
__global__ void f32_to_f16_kernel(const float* __restrict__ src,
                                  _Float16* __restrict__ dst, long n4) {
  long stride = (long)gridDim.x * 256L;
  for (long i = blockIdx.x * 256L + threadIdx.x; i < n4; i += stride) {
    v4f v = ((const v4f*)src)[i];
    ((v4h*)dst)[i] = __builtin_convertvector(v, v4h);
  }
}

// ---------------------------------------------------------------------------
// WMMA GEMM:  C[M,N] = act( alpha * (A[M,K] @ B[N,K]^T) + bias[N] )
// Block tile 128x256x32, 8 waves (2M x 4N), wave tile 64x64
//   -> 16 v_wmma_f32_16x16x32_f16 per wave per K-step vs 16 ds_load_b128.
// Double-buffered LDS fed by the Tensor Data Mover (wave 0 issues two tile
// descriptors per K-step, TENSORcnt), one barrier per K-step.
// LDS row stride 40 halves (80B) -> conflict-free ds_load_b128 across 16 rows.
// ---------------------------------------------------------------------------
template <typename OutT, bool RELU>
__global__ __launch_bounds__(256)
void gemm_ternary(const _Float16* __restrict__ A,
                  const _Float16* __restrict__ B,
                  const float* __restrict__ bias,
                  const float* __restrict__ msum,
                  const unsigned* __restrict__ mcnt,
                  OutT* __restrict__ C,
                  int M, int Nn, int K) {
  constexpr int BM = 128, BN = 256, BK = 32, LD = 40;
  __shared__ _Float16 As[2][BM * LD];
  __shared__ _Float16 Bs[2][BN * LD];

  const int tid  = threadIdx.x;
  const int bm   = blockIdx.y * BM;
  const int bn   = blockIdx.x * BN;
  const int wv   = tid >> 5;
  const int lane = tid & 31;
  const int waveM = (wv & 1) * 64;   // 2 wave rows of 64
  const int waveN = (wv >> 1) * 64;  // 4 wave cols of 64
  const int lhalf = lane >> 4;       // 0 | 1
  const int lmod  = lane & 15;

  // issue one K-step's tile loads into buffer `buf`
  auto stage = [&](int kb, int buf) {
#if USE_TDM
    if (wv == 0) {
      tdm_load_tile(A + (long)bm * K + kb, &As[buf][0], BM, K);
      tdm_load_tile(B + (long)bn * K + kb, &Bs[buf][0], BN, K);
    }
#else
#pragma unroll
    for (int i = 0; i < 2; ++i) {          // A: 512 chunks of 8 halves
      int c = tid + i * 256;
      int row = c >> 2, off = (c & 3) * 8;
      async_load_b128(A + (long)(bm + row) * K + kb + off,
                      &As[buf][row * LD + off]);
    }
#pragma unroll
    for (int i = 0; i < 4; ++i) {          // B: 1024 chunks of 8 halves
      int c = tid + i * 256;
      int row = c >> 2, off = (c & 3) * 8;
      async_load_b128(B + (long)(bn + row) * K + kb + off,
                      &Bs[buf][row * LD + off]);
    }
#endif
  };

  v8f acc[4][4] = {};
  const int nk = K / BK;

  stage(0, 0);

  for (int k = 0; k < nk; ++k) {
    const int buf = k & 1;
#if USE_TDM
    if (wv == 0) wait_tensor0();   // issuing wave: tile k DMA complete
#else
    wait_async0();
#endif
    __syncthreads();               // tile k visible to all waves
    if (k + 1 < nk) stage((k + 1) * BK, buf ^ 1);

    // ---- fragment fetch per documented 16-bit A/B VGPR layouts -------------
    union FH { v16h v; v8h h[2]; };
    FH fa[4], fb[4];
#pragma unroll
    for (int tm = 0; tm < 4; ++tm) {
      const _Float16* p = &As[buf][(waveM + tm * 16 + lmod) * LD];
      fa[tm].h[0] = *(const v8h*)(p + lhalf * 8);       // K 0..7  | 8..15
      fa[tm].h[1] = *(const v8h*)(p + 16 + lhalf * 8);  // K 16..23| 24..31
    }
#pragma unroll
    for (int tn = 0; tn < 4; ++tn) {
      const _Float16* p = &Bs[buf][(waveN + tn * 16 + lmod) * LD];
      fb[tn].h[0] = *(const v8h*)(p + lhalf * 8);       // K 0..7  | 8..15
      fb[tn].h[1] = *(const v8h*)(p + 16 + lhalf * 8);  // K 16..23| 24..31
    }

    // ---- 16 WMMAs ----------------------------------------------------------
#pragma unroll
    for (int tm = 0; tm < 4; ++tm)
#pragma unroll
      for (int tn = 0; tn < 4; ++tn)
        acc[tm][tn] = __builtin_amdgcn_wmma_f32_16x16x32_f16(
            false, fa[tm].v, false, fb[tn].v,
            (short)0, acc[tm][tn], false, false);
  }

  // ---- epilogue: alpha scale + bias + optional ReLU ------------------------
  const float alpha = (*msum) / fmaxf((float)(*mcnt), 1.0f);
#pragma unroll
  for (int tm = 0; tm < 4; ++tm) {
#pragma unroll
    for (int tn = 0; tn < 4; ++tn) {
      int n = bn + waveN + tn * 16 + lmod;
      float bv = bias[n];
#pragma unroll
      for (int r = 0; r < 8; ++r) {
        int m = bm + waveM + tm * 16 + r + 8 * lhalf;
        float v = alpha * acc[tm][tn][r] + bv;
        if (RELU) v = fmaxf(v, 0.f);
        C[(long)m * Nn + n] = (OutT)v;
      }
    }
  }
}

// ---------------------------------------------------------------------------
extern "C" void kernel_launch(void* const* d_in, const int* in_sizes, int n_in,
                              void* d_out, int out_size, void* d_ws, size_t ws_size,
                              hipStream_t stream) {
  const float* x  = (const float*)d_in[0];
  const float* w1 = (const float*)d_in[1];
  const float* b1 = (const float*)d_in[2];
  const float* w2 = (const float*)d_in[3];
  const float* b2 = (const float*)d_in[4];
  const float* w3 = (const float*)d_in[5];
  const float* b3 = (const float*)d_in[6];
  float* out = (float*)d_out;

  char* ws = (char*)d_ws;
  float*    scal = (float*)ws;           // [0]=sumabs1 [1]=msum1 [2]=sumabs2 [3]=msum2 [4]=sumabs3 [5]=msum3
  unsigned* cnts = (unsigned*)(ws + 64); // [0..2] masked counts
  char* p = ws + 256;
  _Float16* wq1 = (_Float16*)p; p += (size_t)DIM_H   * DIM_IN * 2;
  _Float16* wq2 = (_Float16*)p; p += (size_t)DIM_H   * DIM_H  * 2;
  _Float16* wq3 = (_Float16*)p; p += (size_t)DIM_OUT * DIM_H  * 2;
  _Float16* xh  = (_Float16*)p; p += (size_t)DIM_N   * DIM_IN * 2;
  _Float16* h1  = (_Float16*)p; p += (size_t)DIM_N   * DIM_H  * 2;
  _Float16* h2  = (_Float16*)p; p += (size_t)DIM_N   * DIM_H  * 2;

  (void)hipMemsetAsync(ws, 0, 256, stream);

  const long n1 = (long)DIM_H   * DIM_IN;
  const long n2 = (long)DIM_H   * DIM_H;
  const long n3 = (long)DIM_OUT * DIM_H;

  absum_kernel<<<2048, 256, 0, stream>>>(w1, n1, &scal[0]);
  absum_kernel<<<2048, 256, 0, stream>>>(w2, n2, &scal[2]);
  absum_kernel<<<2048, 256, 0, stream>>>(w3, n3, &scal[4]);

  quantize_kernel<<<2048, 256, 0, stream>>>(w1, n1, &scal[0], wq1, &scal[1], &cnts[0]);
  quantize_kernel<<<2048, 256, 0, stream>>>(w2, n2, &scal[2], wq2, &scal[3], &cnts[1]);
  quantize_kernel<<<2048, 256, 0, stream>>>(w3, n3, &scal[4], wq3, &scal[5], &cnts[2]);

  f32_to_f16_kernel<<<2048, 256, 0, stream>>>(x, xh, ((long)DIM_N * DIM_IN) / 4);

  // Layer 1: h1 = relu(alpha1*(xh @ wq1^T) + b1)
  gemm_ternary<_Float16, true>
      <<<dim3(DIM_H / 256, DIM_N / 128), 256, 0, stream>>>(
          xh, wq1, b1, &scal[1], &cnts[0], h1, DIM_N, DIM_H, DIM_IN);

  // Layer 2: h2 = relu(alpha2*(h1 @ wq2^T) + b2)
  gemm_ternary<_Float16, true>
      <<<dim3(DIM_H / 256, DIM_N / 128), 256, 0, stream>>>(
          h1, wq2, b2, &scal[3], &cnts[1], h2, DIM_N, DIM_H, DIM_H);

  // Layer 3: out = alpha3*(h2 @ wq3^T) + b3   (f32 output)
  gemm_ternary<float, false>
      <<<dim3(DIM_OUT / 256, DIM_N / 128), 256, 0, stream>>>(
          h2, wq3, b3, &scal[5], &cnts[2], out, DIM_N, DIM_OUT, DIM_H);
}
